// ResidualQuantizer_36764920054253
// MI455X (gfx1250) — compile-verified, matrix operand order checked by reading, not run on hardware
//
#include <hip/hip_runtime.h>

// ---------------------------------------------------------------------------
// Residual quantizer (RVQ), K=4 stages, C=1024 centroids, D=64, N=131072 rows.
// Distance scoring on v_wmma_f32_16x16x32_bf16 (bf16 A/B, fp32 acc);
// residual recurrence / gather / loss in exact fp32.
// Codebook chunks are double-buffered in LDS and software-pipelined so the
// L2 staging latency hides under the WMMA + argmin compute.
// ---------------------------------------------------------------------------

#define NROWS   (64 * 2048)   // B*T = 131072
#define DIM     64
#define KSTAGES 4
#define CENTS   1024

// output layout (flat concat of reference tuple, all float32)
#define OFF_QUANT 0
#define OFF_LOSS  (NROWS * DIM)                    // 8388608
#define OFF_NN    (OFF_LOSS + NROWS)               // + 131072
#define OFF_CB    (OFF_NN + KSTAGES * NROWS)       // + 524288
#define OFF_CNT   (OFF_CB + KSTAGES * CENTS * DIM) // + 262144

typedef __attribute__((ext_vector_type(16))) __bf16          bf16x16;
typedef __attribute__((ext_vector_type(8)))  float           f32x8;
typedef __attribute__((ext_vector_type(4)))  unsigned short  u16x4;
typedef __attribute__((ext_vector_type(8)))  unsigned short  u16x8;
typedef __attribute__((ext_vector_type(16))) unsigned short  u16x16;

__device__ __forceinline__ unsigned short f32_to_bf16(float f) {
  unsigned int u = __float_as_uint(f);
  u += 0x7fffu + ((u >> 16) & 1u);          // round-to-nearest-even
  return (unsigned short)(u >> 16);
}

// ---------------------------------------------------------------------------
// Prep: zero loss accumulators + counts, precompute fp32 ||c||^2 into ws.
// Re-run every call so graph replays are deterministic.
// ---------------------------------------------------------------------------
__global__ void rvq_prep(const float* __restrict__ cb,
                         float* __restrict__ countsOut,
                         float* __restrict__ ws) {
  int gid = blockIdx.x * blockDim.x + threadIdx.x;
  if (gid < KSTAGES) ws[gid] = 0.0f;                 // loss accumulators
  if (gid < KSTAGES * CENTS) {
    countsOut[gid] = 0.0f;
    const float* row = cb + gid * DIM;
    float s = 0.0f;
#pragma unroll
    for (int i = 0; i < DIM; ++i) { float v = row[i]; s += v * v; }
    ws[KSTAGES + gid] = s;                           // ||c||^2 table
  }
}

// ---------------------------------------------------------------------------
// Main RVQ kernel: 1024 blocks x 256 threads; each wave owns 16 rows.
// ---------------------------------------------------------------------------
__global__ __launch_bounds__(256) void rvq_main(
    const float* __restrict__ in, const float* __restrict__ cb,
    float* __restrict__ outQ, float* __restrict__ outNN,
    float* __restrict__ outCnt, float* __restrict__ wsLoss,
    const float* __restrict__ wsNorm) {
  // double-buffered bf16 codebook chunk (2 x 128 centroids x 64) = 32 KB
  __shared__ __attribute__((aligned(16))) unsigned short cbChunk[2][128 * DIM];
  __shared__ float cnormS[2][128];
  __shared__ int   idxS[8 * 16];   // per-wave argmin indices

  const int tid  = threadIdx.x;
  const int wave = tid >> 5;
  const int lane = tid & 31;
  const int h    = lane >> 4;      // half-wave (selects K-column interleave)
  const int m    = lane & 15;      // A-row / B-column within tile
  const int rowBase = blockIdx.x * 128 + wave * 16;
  const int myRow   = rowBase + m;

  // Residual registers in the WMMA 16-bit A-fragment column pattern:
  // run r8 = s*2 + g covers columns 32*s + 16*g + 8*h + [0..7]  (fp32, exact).
  float res[4][8];
#pragma unroll
  for (int s = 0; s < 2; ++s)
#pragma unroll
    for (int g = 0; g < 2; ++g) {
      const float* p = in + myRow * DIM + 32 * s + 16 * g + 8 * h;
      float4 a = *(const float4*)p;
      float4 b = *(const float4*)(p + 4);
      const int r8 = s * 2 + g;
      res[r8][0] = a.x; res[r8][1] = a.y; res[r8][2] = a.z; res[r8][3] = a.w;
      res[r8][4] = b.x; res[r8][5] = b.y; res[r8][6] = b.z; res[r8][7] = b.w;
    }

  for (int k = 0; k < KSTAGES; ++k) {
    // Build bf16 A fragments (one per 32-wide K step) from the fp32 residual.
    bf16x16 afrag[2];
#pragma unroll
    for (int s = 0; s < 2; ++s) {
      u16x16 aw;
#pragma unroll
      for (int t = 0; t < 8; ++t) {
        aw[t]     = f32_to_bf16(res[s * 2 + 0][t]);
        aw[t + 8] = f32_to_bf16(res[s * 2 + 1][t]);
      }
      afrag[s] = __builtin_bit_cast(bf16x16, aw);
    }

    float minV[8];
    int   minI[8];
#pragma unroll
    for (int r = 0; r < 8; ++r) { minV[r] = 3.4e38f; minI[r] = 0; }

    // ---- software-pipelined, double-buffered chunk loop --------------------
    // prefetch chunk 0 into registers (8 x float4 per thread = 32 KB / block)
    float4 pf[8];
    {
      const float* src = cb + (k * CENTS) * DIM;
#pragma unroll
      for (int j = 0; j < 8; ++j)
        pf[j] = *(const float4*)(src + (j * 256 + tid) * 4);
    }
    float cnPf = (tid < 128) ? wsNorm[k * CENTS + tid] : 0.0f;

    for (int cc = 0; cc < CENTS / 128; ++cc) {       // 8 codebook chunks
      const int buf = cc & 1;
      // convert prefetched fp32 -> bf16, store to LDS (ds_store_b64 quads)
#pragma unroll
      for (int j = 0; j < 8; ++j) {
        u16x4 w;
        w[0] = f32_to_bf16(pf[j].x);
        w[1] = f32_to_bf16(pf[j].y);
        w[2] = f32_to_bf16(pf[j].z);
        w[3] = f32_to_bf16(pf[j].w);
        *(u16x4*)&cbChunk[buf][(j * 256 + tid) * 4] = w;
      }
      if (tid < 128) cnormS[buf][tid] = cnPf;
      __syncthreads();

      // issue next chunk's global loads now; s_wait happens at next iteration
      if (cc < CENTS / 128 - 1) {
        const float* nsrc = cb + (k * CENTS + (cc + 1) * 128) * DIM;
#pragma unroll
        for (int j = 0; j < 8; ++j)
          pf[j] = *(const float4*)(nsrc + (j * 256 + tid) * 4);
        if (tid < 128) cnPf = wsNorm[k * CENTS + (cc + 1) * 128 + tid];
      }

#pragma unroll
      for (int tt = 0; tt < 8; ++tt) {               // 16-centroid tiles
        const int lBase = tt * 16;
        f32x8 acc = {};
#pragma unroll
        for (int s = 0; s < 2; ++s) {                // K = 0..31, 32..63
          const unsigned short* bp =
              &cbChunk[buf][(lBase + m) * DIM + 32 * s + 8 * h];
          u16x8 lo = *(const u16x8*)bp;
          u16x8 hi = *(const u16x8*)(bp + 16);
          u16x16 w = __builtin_shufflevector(lo, hi, 0, 1, 2, 3, 4, 5, 6, 7,
                                             8, 9, 10, 11, 12, 13, 14, 15);
          bf16x16 bfrag = __builtin_bit_cast(bf16x16, w);
          acc = __builtin_amdgcn_wmma_f32_16x16x32_bf16(
              false, afrag[s], false, bfrag, (short)0, acc, false, false);
        }
        const float cn  = cnormS[buf][lBase + m];
        const int   cid = cc * 128 + lBase + m;
#pragma unroll
        for (int r = 0; r < 8; ++r) {
          float sc = __builtin_fmaf(-2.0f, acc[r], cn);  // ||c||^2 - 2 r.c
          if (sc < minV[r]) { minV[r] = sc; minI[r] = cid; }
        }
      }
    }

    // argmin across the 16 lanes of each half-wave (rows r+8h)
#pragma unroll
    for (int mask = 1; mask <= 8; mask <<= 1) {
#pragma unroll
      for (int r = 0; r < 8; ++r) {
        float ov = __shfl_xor(minV[r], mask, 32);
        int   oi = __shfl_xor(minI[r], mask, 32);
        if (ov < minV[r] || (ov == minV[r] && oi < minI[r])) {
          minV[r] = ov; minI[r] = oi;
        }
      }
    }
    if (m == 0) {                                    // lanes 0 and 16
#pragma unroll
      for (int r = 0; r < 8; ++r) idxS[wave * 16 + h * 8 + r] = minI[r];
    }
    __syncthreads();
    const int myIdx = idxS[wave * 16 + m];

    if (lane < 16) {
      outNN[k * NROWS + myRow] = (float)myIdx;
      atomicAdd(&outCnt[k * CENTS + myIdx], 1.0f);
    }

    // gather fp32 centroid (L2-resident), update residual, accumulate loss
    const float* crow = cb + (k * CENTS + myIdx) * DIM;
    float lsum = 0.0f;
#pragma unroll
    for (int s = 0; s < 2; ++s)
#pragma unroll
      for (int g = 0; g < 2; ++g) {
        const float* p = crow + 32 * s + 16 * g + 8 * h;
        float4 a = *(const float4*)p;
        float4 b = *(const float4*)(p + 4);
        const int r8 = s * 2 + g;
        res[r8][0] -= a.x; res[r8][1] -= a.y; res[r8][2] -= a.z; res[r8][3] -= a.w;
        res[r8][4] -= b.x; res[r8][5] -= b.y; res[r8][6] -= b.z; res[r8][7] -= b.w;
#pragma unroll
        for (int e = 0; e < 8; ++e) lsum += res[r8][e] * res[r8][e];
      }
#pragma unroll
    for (int mask = 1; mask < 32; mask <<= 1) lsum += __shfl_xor(lsum, mask, 32);
    if (lane == 0) atomicAdd(&wsLoss[k], lsum);
  }

  // quantized = input - final residual (same column pattern, float4 stores)
#pragma unroll
  for (int s = 0; s < 2; ++s)
#pragma unroll
    for (int g = 0; g < 2; ++g) {
      const int base = myRow * DIM + 32 * s + 16 * g + 8 * h;
      float4 a = *(const float4*)(in + base);
      float4 b = *(const float4*)(in + base + 4);
      const int r8 = s * 2 + g;
      float4 o0 = make_float4(a.x - res[r8][0], a.y - res[r8][1],
                              a.z - res[r8][2], a.w - res[r8][3]);
      float4 o1 = make_float4(b.x - res[r8][4], b.y - res[r8][5],
                              b.z - res[r8][6], b.w - res[r8][7]);
      *(float4*)(outQ + base)     = o0;
      *(float4*)(outQ + base + 4) = o1;
    }
}

// ---------------------------------------------------------------------------
// Finalize: broadcast total loss, copy codebooks out.
// ---------------------------------------------------------------------------
__global__ void rvq_final(const float* __restrict__ cb,
                          const float* __restrict__ wsLoss,
                          float* __restrict__ outLoss,
                          float* __restrict__ outCB) {
  int gid = blockIdx.x * blockDim.x + threadIdx.x;
  if (gid < KSTAGES * CENTS * DIM) outCB[gid] = cb[gid];
  if (gid < NROWS) {
    float t = (wsLoss[0] + wsLoss[1] + wsLoss[2] + wsLoss[3]) *
              (1.0f / (float)(NROWS * DIM));
    outLoss[gid] = t;
  }
}

extern "C" void kernel_launch(void* const* d_in, const int* in_sizes, int n_in,
                              void* d_out, int out_size, void* d_ws, size_t ws_size,
                              hipStream_t stream) {
  (void)in_sizes; (void)n_in; (void)out_size; (void)ws_size;
  const float* in = (const float*)d_in[0];   // [B,T,D] fp32
  const float* cb = (const float*)d_in[1];   // [K,C,D] fp32

  float* out     = (float*)d_out;
  float* outQ    = out + OFF_QUANT;
  float* outLoss = out + OFF_LOSS;
  float* outNN   = out + OFF_NN;
  float* outCB   = out + OFF_CB;
  float* outCnt  = out + OFF_CNT;

  float* ws     = (float*)d_ws;              // [0..3] loss, [4..4+4096) norms
  float* wsLoss = ws;
  float* wsNorm = ws + KSTAGES;

  rvq_prep<<<16, 256, 0, stream>>>(cb, outCnt, ws);
  rvq_main<<<NROWS / 128, 256, 0, stream>>>(in, cb, outQ, outNN, outCnt,
                                            wsLoss, wsNorm);
  rvq_final<<<(KSTAGES * CENTS * DIM + 255) / 256, 256, 0, stream>>>(
      cb, wsLoss, outLoss, outCB);
}
